// MambaLikeLM_47467978555452
// MI455X (gfx1250) — matrix operand order repaired and checked by reading, not compiled, not run on hardware
//
#include <hip/hip_runtime.h>
#include <hip/hip_bf16.h>
#include <math.h>

#define V_    32000
#define D_    512
#define NL_   6
#define B_    2
#define S_    2048
#define H_    8
#define N_    16
#define KC_   4
#define DI_   1024
#define TOT_  2312          // DI + DI + H + H*N + H*N
#define ROWS_ (B_*S_)       // 4096

typedef __attribute__((ext_vector_type(16))) __bf16 v16bf;
typedef __attribute__((ext_vector_type(8)))  __bf16 v8bf;
typedef __attribute__((ext_vector_type(8)))  float  v8f;
typedef __attribute__((__vector_size__(16))) int    v4i;

// --- CDNA5 async global->LDS path (bypasses VGPRs, tracked by ASYNCcnt) ----
#ifndef __has_builtin
#define __has_builtin(x) 0
#endif
#if __has_builtin(__builtin_amdgcn_global_load_async_to_lds_b128)
#define USE_ASYNC_LDS 1
#else
#define USE_ASYNC_LDS 0
#endif

#if USE_ASYNC_LDS
__device__ __forceinline__ void async_cp16(const void* gsrc, void* ldst) {
    __builtin_amdgcn_global_load_async_to_lds_b128(
        (__attribute__((address_space(1))) v4i*)gsrc,
        (__attribute__((address_space(3))) v4i*)ldst, 0, 0);
}
#endif

__device__ __forceinline__ void async_wait0() {
#if USE_ASYNC_LDS
#if __has_builtin(__builtin_amdgcn_s_wait_asynccnt)
    __builtin_amdgcn_s_wait_asynccnt(0);
#else
    asm volatile("s_wait_asynccnt 0x0" ::: "memory");
#endif
#endif
}

// ---------------------------------------------------------------------------
// fp32 -> bf16 streaming convert (weights + activations)
// ---------------------------------------------------------------------------
__global__ void f32_to_bf16_kernel(const float* __restrict__ in,
                                   __bf16* __restrict__ out, long long n) {
    long long i = (long long)blockIdx.x * blockDim.x + threadIdx.x;
    long long stride = (long long)gridDim.x * blockDim.x;
    for (; i < n; i += stride) out[i] = (__bf16)in[i];
}

// ---------------------------------------------------------------------------
// embedding gather: x[row, :] = embed[tokens[row], :]
// ---------------------------------------------------------------------------
__global__ __launch_bounds__(256)
void embed_kernel(const int* __restrict__ tokens, const float* __restrict__ embed,
                  float* __restrict__ x) {
    int row = blockIdx.x;
    int tok = tokens[row];
    const float* src = embed + (size_t)tok * D_;
    float* dst = x + (size_t)row * D_;
    for (int d = threadIdx.x; d < D_; d += 256) dst[d] = src[d];
}

// ---------------------------------------------------------------------------
// RMSNorm -> bf16
// ---------------------------------------------------------------------------
__global__ __launch_bounds__(256)
void rmsnorm_kernel(const float* __restrict__ x, const float* __restrict__ w,
                    __bf16* __restrict__ out) {
    __shared__ float red[256];
    int row = blockIdx.x;
    const float* xr = x + (size_t)row * D_;
    float s = 0.f;
    for (int d = threadIdx.x; d < D_; d += 256) { float v = xr[d]; s += v * v; }
    red[threadIdx.x] = s;
    __syncthreads();
    for (int st = 128; st > 0; st >>= 1) {
        if (threadIdx.x < st) red[threadIdx.x] += red[threadIdx.x + st];
        __syncthreads();
    }
    float rinv = rsqrtf(red[0] * (1.f / D_) + 1e-6f);
    for (int d = threadIdx.x; d < D_; d += 256)
        out[(size_t)row * D_ + d] = (__bf16)(xr[d] * rinv * w[d]);
}

// ---------------------------------------------------------------------------
// Final LayerNorm -> bf16
// ---------------------------------------------------------------------------
__global__ __launch_bounds__(256)
void layernorm_kernel(const float* __restrict__ x, const float* __restrict__ w,
                      const float* __restrict__ b, __bf16* __restrict__ out) {
    __shared__ float r1[256];
    __shared__ float r2[256];
    int row = blockIdx.x;
    const float* xr = x + (size_t)row * D_;
    float s1 = 0.f, s2 = 0.f;
    for (int d = threadIdx.x; d < D_; d += 256) { float v = xr[d]; s1 += v; s2 += v * v; }
    r1[threadIdx.x] = s1; r2[threadIdx.x] = s2;
    __syncthreads();
    for (int st = 128; st > 0; st >>= 1) {
        if (threadIdx.x < st) { r1[threadIdx.x] += r1[threadIdx.x + st];
                                r2[threadIdx.x] += r2[threadIdx.x + st]; }
        __syncthreads();
    }
    float mu  = r1[0] * (1.f / D_);
    float var = r2[0] * (1.f / D_) - mu * mu;
    float rstd = rsqrtf(var + 1e-5f);
    for (int d = threadIdx.x; d < D_; d += 256)
        out[(size_t)row * D_ + d] = (__bf16)((xr[d] - mu) * rstd * w[d] + b[d]);
}

// ---------------------------------------------------------------------------
// bf16 WMMA GEMM:  C[M,N] = A[M,K] * W[N,K]^T (+bias[N]) (+resid[M,N])
// 128x128 block, 4 waves, each wave 64x64 via 4x4 v_wmma_f32_16x16x32_bf16.
// Double-buffered LDS tiles; next K-tile streams in via async global->LDS
// while WMMAs run on the current tile.
// ---------------------------------------------------------------------------
__global__ __launch_bounds__(128)
void gemm_bf16_wmma(const __bf16* __restrict__ A, int lda,
                    const __bf16* __restrict__ W, int ldw,
                    float* __restrict__ C, long long ldc,
                    const float* __restrict__ bias,
                    const float* __restrict__ resid, int ldr,
                    int M, int Nn, int Kd) {
    __shared__ __align__(16) __bf16 As[2][128][40];
    __shared__ __align__(16) __bf16 Ws[2][128][40];

    const int tid  = threadIdx.x;
    const int lane = tid & 31;
    const int wave = tid >> 5;
    const int m0 = blockIdx.y * 128;
    const int n0 = blockIdx.x * 128;
    const int wm = (wave >> 1) * 64;
    const int wn = (wave & 1) * 64;
    const int h = lane >> 4;    // half-wave select
    const int r = lane & 15;    // row/col within 16

    v8f acc[4][4];
    #pragma unroll
    for (int i = 0; i < 4; i++)
        #pragma unroll
        for (int j = 0; j < 4; j++)
            acc[i][j] = (v8f){0.f,0.f,0.f,0.f,0.f,0.f,0.f,0.f};

    const int wrow = (n0 + tid < Nn) ? (n0 + tid) : (Nn - 1);  // clamp OOB weight rows
    const __bf16* arow = A + (size_t)(m0 + tid) * lda;
    const __bf16* wrp  = W + (size_t)wrow * ldw;

    // stage one 128x32 A tile and one 128x32 W tile into LDS buffer `bsel`
    auto stage = [&](int bsel, int k0) {
#if USE_ASYNC_LDS
        #pragma unroll
        for (int c = 0; c < 4; c++) {
            async_cp16(arow + k0 + c * 8, &As[bsel][tid][c * 8]);
            async_cp16(wrp  + k0 + c * 8, &Ws[bsel][tid][c * 8]);
        }
#else
        const uint4* ga = (const uint4*)(arow + k0);
        const uint4* gw = (const uint4*)(wrp + k0);
        #pragma unroll
        for (int c = 0; c < 4; c++) {
            *(uint4*)&As[bsel][tid][c * 8] = ga[c];
            *(uint4*)&Ws[bsel][tid][c * 8] = gw[c];
        }
#endif
    };

    stage(0, 0);
    int bsel = 0;
    for (int k0 = 0; k0 < Kd; k0 += 32) {
        async_wait0();        // this wave's tile loads landed in LDS
        __syncthreads();      // ... and everyone else's too

        if (k0 + 32 < Kd) stage(bsel ^ 1, k0 + 32);   // overlap next tile with compute

        // A fragments: lane holds row (r), K = {h*8..h*8+7, 16+h*8..16+h*8+7}
        v16bf afr[4], bfr[4];
        #pragma unroll
        for (int i = 0; i < 4; i++) {
            const __bf16* ap = &As[bsel][wm + i * 16 + r][0];
            v8bf c0 = *(const v8bf*)(ap + h * 8);
            v8bf c1 = *(const v8bf*)(ap + 16 + h * 8);
            afr[i] = __builtin_shufflevector(c0, c1, 0,1,2,3,4,5,6,7,8,9,10,11,12,13,14,15);
        }
        // B fragments: lane holds col (r), K = {h*8..h*8+7, 16+h*8..16+h*8+7}
        #pragma unroll
        for (int j = 0; j < 4; j++) {
            const __bf16* bp = &Ws[bsel][wn + j * 16 + r][0];
            v8bf c0 = *(const v8bf*)(bp + h * 8);
            v8bf c1 = *(const v8bf*)(bp + 16 + h * 8);
            bfr[j] = __builtin_shufflevector(c0, c1, 0,1,2,3,4,5,6,7,8,9,10,11,12,13,14,15);
        }
        #pragma unroll
        for (int i = 0; i < 4; i++)
            #pragma unroll
            for (int j = 0; j < 4; j++)
                acc[i][j] = __builtin_amdgcn_wmma_f32_16x16x32_bf16(
                    false, afr[i], false, bfr[j], (short)0, acc[i][j], false, false);

        __syncthreads();      // all waves done reading buffer before it is reused
        bsel ^= 1;
    }

    // D layout: lane -> n = r, vgpr rr -> m = rr + 8*h
    #pragma unroll
    for (int j = 0; j < 4; j++) {
        int n = n0 + wn + j * 16 + r;
        if (n >= Nn) continue;
        float bv = bias ? bias[n] : 0.f;
        #pragma unroll
        for (int i = 0; i < 4; i++) {
            #pragma unroll
            for (int rr = 0; rr < 8; rr++) {
                int m = m0 + wm + i * 16 + h * 8 + rr;
                float v = acc[i][j][rr] + bv;
                if (resid) v += resid[(size_t)m * ldr + n];
                C[(size_t)m * ldc + n] = v;
            }
        }
    }
}

// ---------------------------------------------------------------------------
// causal depthwise conv (K=4) + SiLU over xi channels of p
// ---------------------------------------------------------------------------
__global__ __launch_bounds__(256)
void conv_silu_kernel(const float* __restrict__ p, const float* __restrict__ cw,
                      float* __restrict__ xi) {
    int c   = blockIdx.x * 256 + threadIdx.x;   // 0..DI-1
    int row = blockIdx.y;                        // 0..ROWS-1
    int b = row >> 11;                           // S_ = 2048
    int s = row & (S_ - 1);
    float w0 = cw[c * KC_ + 0], w1 = cw[c * KC_ + 1];
    float w2 = cw[c * KC_ + 2], w3 = cw[c * KC_ + 3];
    const float* pb = p + (size_t)b * S_ * TOT_ + DI_ + c;
    float acc = 0.f;
    if (s - 3 >= 0) acc += pb[(size_t)(s - 3) * TOT_] * w0;
    if (s - 2 >= 0) acc += pb[(size_t)(s - 2) * TOT_] * w1;
    if (s - 1 >= 0) acc += pb[(size_t)(s - 1) * TOT_] * w2;
    acc += pb[(size_t)s * TOT_] * w3;
    xi[(size_t)row * DI_ + c] = acc / (1.f + expf(-acc));   // silu
}

// ---------------------------------------------------------------------------
// per-(row,head): x_in mean (128 ch), dt=softplus(p_dt+bias), dA=exp(dt*A),
// u[n] = dt * x_in * B[n].  128 threads = 8 heads x 16 lanes.
// ---------------------------------------------------------------------------
__global__ __launch_bounds__(128)
void prep_kernel(const float* __restrict__ p, const float* __restrict__ xi,
                 const float* __restrict__ dtb, const float* __restrict__ alog,
                 float* __restrict__ dAbuf, float* __restrict__ ubuf) {
    int row  = blockIdx.x;
    int t    = threadIdx.x;
    int head = t >> 4;
    int ln   = t & 15;
    const float* xr = xi + (size_t)row * DI_ + head * 128 + ln * 8;
    float s = 0.f;
    #pragma unroll
    for (int q = 0; q < 8; q++) s += xr[q];
    s += __shfl_xor(s, 1, 16);
    s += __shfl_xor(s, 2, 16);
    s += __shfl_xor(s, 4, 16);
    s += __shfl_xor(s, 8, 16);
    float x_in = s * (1.f / 128.f);
    float dtv = p[(size_t)row * TOT_ + 2 * DI_ + head] + dtb[head];
    float dt  = (dtv > 20.f) ? dtv : log1pf(expf(dtv));
    float Ah  = -expf(alog[head]);
    if (ln == 0) dAbuf[(size_t)row * H_ + head] = expf(dt * Ah);
    float Bm = p[(size_t)row * TOT_ + 2 * DI_ + H_ + head * N_ + ln];
    ubuf[((size_t)row * H_ + head) * N_ + ln] = dt * x_in * Bm;
}

// ---------------------------------------------------------------------------
// sequential SSM scan: one block per (b,h); 16 lanes hold the N=16 state
// ---------------------------------------------------------------------------
__global__ __launch_bounds__(32)
void scan_kernel(const float* __restrict__ p, const float* __restrict__ dAbuf,
                 const float* __restrict__ ubuf, float* __restrict__ ybuf) {
    int bh = blockIdx.x;
    int b  = bh / H_;
    int hh = bh % H_;
    int n  = threadIdx.x;
    if (n >= N_) return;
    float st = 0.f;
    for (int t = 0; t < S_; t++) {
        size_t row = (size_t)b * S_ + t;
        float dA = dAbuf[row * H_ + hh];
        float u  = ubuf[(row * H_ + hh) * N_ + n];
        float Cv = p[row * TOT_ + 2 * DI_ + H_ + H_ * N_ + hh * N_ + n];
        st = dA * st + u;
        float pr = st * Cv;
        pr += __shfl_xor(pr, 1, 16);
        pr += __shfl_xor(pr, 2, 16);
        pr += __shfl_xor(pr, 4, 16);
        pr += __shfl_xor(pr, 8, 16);
        if (n == 0) ybuf[row * H_ + hh] = pr;
    }
}

// ---------------------------------------------------------------------------
// y_proj (K=8) + SiLU(z) gating -> bf16 input for out_proj GEMM
// ---------------------------------------------------------------------------
__global__ __launch_bounds__(256)
void yproj_gate_kernel(const float* __restrict__ p, const float* __restrict__ ybuf,
                       const float* __restrict__ yw, __bf16* __restrict__ gated) {
    int i   = blockIdx.x * 256 + threadIdx.x;
    int row = blockIdx.y;
    const float* yr = ybuf + (size_t)row * H_;
    const float* wr = yw + (size_t)i * H_;
    float acc = 0.f;
    #pragma unroll
    for (int hh = 0; hh < H_; hh++) acc += yr[hh] * wr[hh];
    float z = p[(size_t)row * TOT_ + i];
    float sil = z / (1.f + expf(-z));
    gated[(size_t)row * DI_ + i] = (__bf16)(acc * sil);
}

// ---------------------------------------------------------------------------
// host launcher
// ---------------------------------------------------------------------------
extern "C" void kernel_launch(void* const* d_in, const int* in_sizes, int n_in,
                              void* d_out, int out_size, void* d_ws, size_t ws_size,
                              hipStream_t stream) {
    (void)in_sizes; (void)n_in; (void)out_size; (void)ws_size;
    const int*   tokens     = (const int*)  d_in[0];
    const float* embed      = (const float*)d_in[1];
    const float* rms_w      = (const float*)d_in[2];
    const float* in_proj_w  = (const float*)d_in[3];
    const float* conv_w     = (const float*)d_in[4];
    const float* dt_bias    = (const float*)d_in[5];
    const float* A_log      = (const float*)d_in[6];
    const float* y_proj_w   = (const float*)d_in[7];
    const float* out_proj_w = (const float*)d_in[8];
    const float* ln_w       = (const float*)d_in[9];
    const float* ln_b       = (const float*)d_in[10];
    const float* head_w     = (const float*)d_in[11];
    const float* head_b     = (const float*)d_in[12];
    float* out = (float*)d_out;

    char* ws = (char*)d_ws;
    size_t off = 0;
    auto take = [&](size_t bytes) -> char* {
        char* q = ws + off;
        off = (off + bytes + 255) & ~(size_t)255;
        return q;
    };
    float*  x     = (float*) take((size_t)ROWS_ * D_ * 4);
    float*  pbuf  = (float*) take((size_t)ROWS_ * TOT_ * 4);
    float*  xibuf = (float*) take((size_t)ROWS_ * DI_ * 4);
    float*  dAbuf = (float*) take((size_t)ROWS_ * H_ * 4);
    float*  ubuf  = (float*) take((size_t)ROWS_ * H_ * N_ * 4);
    float*  ybuf  = (float*) take((size_t)ROWS_ * H_ * 4);
    __bf16* xnbf  = (__bf16*)take((size_t)ROWS_ * D_ * 2);
    __bf16* gbf   = (__bf16*)take((size_t)ROWS_ * DI_ * 2);
    __bf16* win   = (__bf16*)take((size_t)NL_ * TOT_ * D_ * 2);
    __bf16* wout  = (__bf16*)take((size_t)NL_ * D_ * DI_ * 2);
    __bf16* whead = (__bf16*)take((size_t)V_ * D_ * 2);

    // one-time-per-call weight downconversion (streams through L2)
    f32_to_bf16_kernel<<<2048, 256, 0, stream>>>(in_proj_w,  win,  (long long)NL_ * TOT_ * D_);
    f32_to_bf16_kernel<<<2048, 256, 0, stream>>>(out_proj_w, wout, (long long)NL_ * D_ * DI_);
    f32_to_bf16_kernel<<<4096, 256, 0, stream>>>(head_w,     whead,(long long)V_ * D_);

    embed_kernel<<<ROWS_, 256, 0, stream>>>(tokens, embed, x);

    for (int l = 0; l < NL_; l++) {
        rmsnorm_kernel<<<ROWS_, 256, 0, stream>>>(x, rms_w + (size_t)l * D_, xnbf);
        // p = xn @ W_in^T   (4096 x 2312, K=512)
        gemm_bf16_wmma<<<dim3((TOT_ + 127) / 128, ROWS_ / 128), 128, 0, stream>>>(
            xnbf, D_, win + (size_t)l * TOT_ * D_, D_, pbuf, TOT_,
            nullptr, nullptr, 0, ROWS_, TOT_, D_);
        conv_silu_kernel<<<dim3(DI_ / 256, ROWS_), 256, 0, stream>>>(
            pbuf, conv_w + (size_t)l * DI_ * KC_, xibuf);
        prep_kernel<<<ROWS_, 128, 0, stream>>>(
            pbuf, xibuf, dt_bias + (size_t)l * H_, A_log + (size_t)l * H_, dAbuf, ubuf);
        scan_kernel<<<B_ * H_, 32, 0, stream>>>(pbuf, dAbuf, ubuf, ybuf);
        yproj_gate_kernel<<<dim3(DI_ / 256, ROWS_), 256, 0, stream>>>(
            pbuf, ybuf, y_proj_w + (size_t)l * DI_ * H_, gbf);
        // x = gated @ W_out^T + x   (4096 x 512, K=1024), residual in place
        gemm_bf16_wmma<<<dim3(D_ / 128, ROWS_ / 128), 128, 0, stream>>>(
            gbf, DI_, wout + (size_t)l * D_ * DI_, DI_, x, D_,
            nullptr, x, D_, ROWS_, D_, DI_);
    }

    layernorm_kernel<<<ROWS_, 256, 0, stream>>>(x, ln_w, ln_b, xnbf);
    // logits = ln(x) @ head_w^T + head_b   (4096 x 32000, K=512)
    gemm_bf16_wmma<<<dim3(V_ / 128, ROWS_ / 128), 128, 0, stream>>>(
        xnbf, D_, whead, D_, out, V_, head_b, nullptr, 0, ROWS_, V_, D_);
}
